// SpatialMemoryNet_54150947668394
// MI455X (gfx1250) — matrix operand order repaired and checked by reading, not compiled, not run on hardware
//
#include <hip/hip_runtime.h>
#include <hip/hip_bf16.h>

typedef __bf16 bf16;
typedef __bf16 v8bf  __attribute__((ext_vector_type(8)));
typedef __bf16 v16bf __attribute__((ext_vector_type(16)));
typedef float  v8f   __attribute__((ext_vector_type(8)));
typedef unsigned int u32x4 __attribute__((ext_vector_type(4)));
typedef unsigned int u32x8 __attribute__((ext_vector_type(8)));

union Frag { v16bf v; v8bf h[2]; };

// ---------------- WMMA helpers (CDNA5 wave32) ----------------
__device__ __forceinline__ v8f wmma_bf16(v16bf a, v16bf b, v8f c) {
  return __builtin_amdgcn_wmma_f32_16x16x32_bf16(false, a, false, b, (short)0, c,
                                                 false, false);
}

// A-operand (16x32, row-major tile in LDS): lane holds row (l&15);
// lanes 0-15 hold K=[0..7]+[16..23], lanes 16-31 K=[8..15]+[24..31].
__device__ __forceinline__ v16bf load_a_frag(const bf16* base, int stride, int row0, int k0) {
  int l = (int)(threadIdx.x & 31u), kh = l >> 4, r = l & 15;
  const bf16* p = base + (size_t)(row0 + r) * stride + k0;
  Frag f;
  f.h[0] = *(const v8bf*)(p + kh * 8);
  f.h[1] = *(const v8bf*)(p + 16 + kh * 8);
  return f.v;
}
// B-operand (32x16) from column-major staged tile Bl[col][k]:
// lane holds col (l&15); lanes 0-15 K=[0..7]+[16..23], lanes 16-31 K=[8..15]+[24..31].
__device__ __forceinline__ v16bf load_b_frag(const bf16* base, int stride, int col0, int k0) {
  int l = (int)(threadIdx.x & 31u), kh = l >> 4, c = l & 15;
  const bf16* p = base + (size_t)(col0 + c) * stride + k0;
  Frag f;
  f.h[0] = *(const v8bf*)(p + kh * 8);
  f.h[1] = *(const v8bf*)(p + 16 + kh * 8);
  return f.v;
}

// ---------------- Tensor Data Mover (async tile DMA global->LDS) ----------------
__device__ __forceinline__ void tdm_load_2d(unsigned lds_byte_off, const void* gptr,
                                            unsigned row_elems, unsigned rows,
                                            unsigned stride_elems,
                                            unsigned pad_i, unsigned pad_a, bool pad) {
  unsigned long long ga = (unsigned long long)gptr;
  u32x4 g0; u32x8 g1;
  g0[0] = 1u;                                   // count=1, user descriptor
  g0[1] = lds_byte_off;                         // lds_addr
  g0[2] = (unsigned)ga;                         // global_addr[31:0]
  g0[3] = ((unsigned)(ga >> 32) & 0x01ffffffu) | (2u << 30);  // addr[56:32] | type=2
  g1[0] = (1u << 16) |                          // data_size = 2B (bf16)
          (pad ? ((1u << 20) | (pad_i << 22) | (pad_a << 25)) : 0u);
  g1[1] = (row_elems & 0xffffu) << 16;          // tensor_dim0 lo16
  g1[2] = ((row_elems >> 16) & 0xffffu) | ((rows & 0xffffu) << 16);   // dim0 hi | dim1 lo
  g1[3] = ((rows >> 16) & 0xffffu) | ((row_elems & 0xffffu) << 16);   // dim1 hi | tile_dim0
  g1[4] = rows & 0xffffu;                       // tile_dim1 (tile_dim2 = 0)
  g1[5] = stride_elems;                         // tensor_dim0_stride lo32
  g1[6] = 0u;
  g1[7] = 0u;
  asm volatile("tensor_load_to_lds %0, %1" :: "s"(g0), "s"(g1) : "memory");
}
__device__ __forceinline__ void tdm_wait0() { __builtin_amdgcn_s_wait_tensorcnt(0); }
__device__ __forceinline__ void tdm_wait1() { __builtin_amdgcn_s_wait_tensorcnt(1); }
__device__ __forceinline__ void tdm_wait2() { __builtin_amdgcn_s_wait_tensorcnt(2); }
__device__ __forceinline__ unsigned lds_off(const void* p, const char* base) {
  return __builtin_amdgcn_groupstaticsize() + (unsigned)((const char*)p - base);
}
// 64x64 bf16 tile from a stride-512 matrix, LDS-padded to stride 72
__device__ __forceinline__ void tdm_load_tile64(unsigned off, const bf16* g) {
  tdm_load_2d(off, g, 64, 64, 512, 4, 3, true);
}

// ---------------- elementwise fp32 -> bf16 ----------------
__global__ void cvt_f32_bf16(const float* __restrict__ s, bf16* __restrict__ d, long n) {
  long i = (long)blockIdx.x * blockDim.x + threadIdx.x;
  long str = (long)gridDim.x * blockDim.x;
  for (; i < n; i += str) d[i] = (bf16)s[i];
}

// ---------------- transpose + convert: Wt[n][k] = bf16(W[k][n]) ----------------
__global__ void cvt_transpose_bf16(const float* __restrict__ s, bf16* __restrict__ d, int D) {
  __shared__ float t[64][65];
  int bk = blockIdx.x * 64, bn = blockIdx.y * 64;
  for (int i = (int)threadIdx.x; i < 64 * 64; i += 256) {
    int k = i >> 6, n = i & 63;
    t[k][n] = s[(size_t)(bk + k) * D + bn + n];
  }
  __syncthreads();
  for (int i = (int)threadIdx.x; i < 64 * 64; i += 256) {
    int n = i >> 6, k = i & 63;
    d[(size_t)(bn + n) * D + bk + k] = (bf16)t[k][n];
  }
}

// ---------------- beta = sigmoid(X @ Wbeta + b) ----------------
__global__ void beta_kernel(const float* __restrict__ X, const float* __restrict__ Wb,
                            const float* __restrict__ bb, float* __restrict__ beta, int D) {
  int wave = (int)(threadIdx.x >> 5), lane = (int)(threadIdx.x & 31u);
  long row = (long)blockIdx.x * 8 + wave;
  const float* x = X + row * D;
  float acc = 0.f;
  for (int k = lane; k < D; k += 32) acc += x[k] * Wb[k];
  for (int off = 16; off > 0; off >>= 1) acc += __shfl_xor(acc, off, 32);
  if (lane == 0) beta[row] = 1.f / (1.f + __expf(-(acc + bb[0])));
}

// ---------------- bf16 WMMA GEMM: out = A[MxK] * Bt[NxK]^T + bias ----------------
// block = 256 threads (8 waves), tile 128x64, K-step 32.
// BOTH tiles are 2-D TDM DMAs (A and pre-transposed weights), double-buffered.
template <bool OUTF32>
__global__ void gemm_wmma(const bf16* __restrict__ A, const bf16* __restrict__ Bt,
                          const float* __restrict__ bias, void* __restrict__ out,
                          int M, int N, int K) {
  extern __shared__ char smem[];
  const int ATILE = 128 * 40, BTILE = 64 * 40;
  bf16* AlB = (bf16*)smem;              // 2 x [128][40]
  bf16* BlB = (bf16*)smem + 2 * ATILE;  // 2 x [64][40]
  const int tid = (int)threadIdx.x;
  const int wave = tid >> 5, wm = wave >> 1, wn = wave & 1;
  const int tm = blockIdx.x * 128, tn = blockIdx.y * 64;
  const int nsteps = K >> 5;
  v8f acc[2][2] = {};
  if (wave == 0) {  // prologue: DMA first A / B^T tiles (row=32 elems -> pad to stride 40)
    tdm_load_2d(lds_off(AlB, smem), A + (size_t)tm * K, 32, 128, (unsigned)K, 3, 3, true);
    tdm_load_2d(lds_off(BlB, smem), Bt + (size_t)tn * K, 32, 64, (unsigned)K, 3, 3, true);
  }
  for (int kt = 0; kt < nsteps; ++kt) {
    const int cur = kt & 1;
    const bf16* Alc = AlB + cur * ATILE;
    const bf16* Blc = BlB + cur * BTILE;
    if (wave == 0) {
      if (kt + 1 < nsteps) {   // prefetch next tiles into alternate buffers
        tdm_load_2d(lds_off(AlB + (cur ^ 1) * ATILE, smem),
                    A + (size_t)tm * K + (kt + 1) * 32, 32, 128, (unsigned)K, 3, 3, true);
        tdm_load_2d(lds_off(BlB + (cur ^ 1) * BTILE, smem),
                    Bt + (size_t)tn * K + (kt + 1) * 32, 32, 64, (unsigned)K, 3, 3, true);
        tdm_wait2();           // current pair complete; only prefetch outstanding
      } else {
        tdm_wait0();
      }
    }
    __syncthreads();
    v16bf a0 = load_a_frag(Alc, 40, wm * 32, 0);
    v16bf a1 = load_a_frag(Alc, 40, wm * 32 + 16, 0);
    v16bf b0 = load_b_frag(Blc, 40, wn * 32, 0);
    v16bf b1 = load_b_frag(Blc, 40, wn * 32 + 16, 0);
    acc[0][0] = wmma_bf16(a0, b0, acc[0][0]);
    acc[0][1] = wmma_bf16(a0, b1, acc[0][1]);
    acc[1][0] = wmma_bf16(a1, b0, acc[1][0]);
    acc[1][1] = wmma_bf16(a1, b1, acc[1][1]);
    __syncthreads();
  }
  int l = tid & 31, kh = l >> 4, l15 = l & 15;
  #pragma unroll
  for (int fi = 0; fi < 2; ++fi)
    #pragma unroll
    for (int fj = 0; fj < 2; ++fj)
      #pragma unroll
      for (int r = 0; r < 8; ++r) {
        int row = tm + wm * 32 + fi * 16 + r + 8 * kh;
        int col = tn + wn * 32 + fj * 16 + l15;
        float v = acc[fi][fj][r] + (bias ? bias[col] : 0.f);
        if (OUTF32) ((float*)out)[(size_t)row * N + col] = v;
        else        ((bf16*)out)[(size_t)row * N + col] = (bf16)v;
      }
}

// ---------------- per-chunk prep: T (fwd-subst), W = T*Kb, U = T*Vb ----------------
__global__ void chunkprep(const bf16* __restrict__ Kp, const bf16* __restrict__ Vp,
                          const float* __restrict__ betap,
                          bf16* __restrict__ Wp, bf16* __restrict__ Up) {
  extern __shared__ char smem[];
  const int KS = 520;
  bf16* Kl = (bf16*)smem;                 // [64][520]
  bf16* Vl = Kl + 64 * KS;                // [64][520]
  float* Tf = (float*)(Vl + 64 * KS);     // [64][65] fp32
  bf16* Tb = (bf16*)(Tf + 64 * 65);       // [64][72]   (T+I) * diag(beta)
  bf16* Bt = Tb + 64 * 72;                // [64][72]
  float* betas = (float*)(Bt + 64 * 72);  // [64]
  const int tid = (int)threadIdx.x;
  const int wave = tid >> 5, wm = wave >> 1, wn = wave & 1;
  const int l = tid & 31, kh = l >> 4, l15 = l & 15;
  const int n = blockIdx.x & 63, b = blockIdx.x >> 6;
  const size_t g0 = (size_t)b * 4096 + (size_t)n * 64;

  if (wave == 0) {  // whole K/V chunk DMA, row=256DW -> pad to stride 520
    tdm_load_2d(lds_off(Kl, smem), Kp + g0 * 512, 512, 64, 512, 7, 3, true);
    tdm_load_2d(lds_off(Vl, smem), Vp + g0 * 512, 512, 64, 512, 7, 3, true);
  }
  if (tid < 64) betas[tid] = betap[g0 + tid];
  if (wave == 0) tdm_wait0();
  __syncthreads();

  // acc = K @ K^T (beta folded into epilogue: T = -diag(beta) * acc, strict lower)
  v8f tacc[2] = {};
  for (int k0 = 0; k0 < 512; k0 += 32) {
    v16bf a  = load_a_frag(Kl, KS, wm * 16, k0);
    v16bf b0 = load_b_frag(Kl, KS, wn * 32, k0);
    v16bf b1 = load_b_frag(Kl, KS, wn * 32 + 16, k0);
    tacc[0] = wmma_bf16(a, b0, tacc[0]);
    tacc[1] = wmma_bf16(a, b1, tacc[1]);
  }
  #pragma unroll
  for (int f = 0; f < 2; ++f)
    #pragma unroll
    for (int r = 0; r < 8; ++r) {
      int row = wm * 16 + r + 8 * kh;
      int col = wn * 32 + f * 16 + l15;
      Tf[row * 65 + col] = (row > col) ? -betas[row] * tacc[f][r] : 0.f;
    }
  __syncthreads();

  // forward substitution (serial over rows, parallel over columns)
  for (int i = 1; i < 64; ++i) {
    float upd = 0.f;
    if (tid < i) {
      for (int m = tid + 1; m < i; ++m) upd += Tf[i * 65 + m] * Tf[m * 65 + tid];
    }
    __syncthreads();
    if (tid < i) Tf[i * 65 + tid] += upd;
    __syncthreads();
  }
  // Tb = bf16((T + I) * diag(beta))  -> W = Tb @ K, U = Tb @ V
  for (int i = tid; i < 64 * 64; i += 256) {
    int r = i >> 6, c = i & 63;
    Tb[r * 72 + c] = (bf16)((Tf[r * 65 + c] + (r == c ? 1.f : 0.f)) * betas[c]);
  }
  __syncthreads();

  for (int nt = 0; nt < 8; ++nt) {
    for (int pass = 0; pass < 2; ++pass) {
      const bf16* src = pass ? Vl : Kl;
      bf16* dstg = pass ? Up : Wp;
      for (int i = tid; i < 64 * 64; i += 256) {   // pure transpose: Bt[v][c] = src[c][nt*64+v]
        int c = i >> 6, v = i & 63;
        Bt[v * 72 + c] = src[(size_t)c * KS + nt * 64 + v];
      }
      __syncthreads();
      v8f acc[2] = {};
      #pragma unroll
      for (int k0 = 0; k0 < 64; k0 += 32) {
        v16bf a  = load_a_frag(Tb, 72, wm * 16, k0);
        v16bf b0 = load_b_frag(Bt, 72, wn * 32, k0);
        v16bf b1 = load_b_frag(Bt, 72, wn * 32 + 16, k0);
        acc[0] = wmma_bf16(a, b0, acc[0]);
        acc[1] = wmma_bf16(a, b1, acc[1]);
      }
      #pragma unroll
      for (int f = 0; f < 2; ++f)
        #pragma unroll
        for (int r = 0; r < 8; ++r) {
          int row = wm * 16 + r + 8 * kh;
          int col = wn * 32 + f * 16 + l15;
          dstg[(g0 + row) * 512 + nt * 64 + col] = (bf16)acc[f][r];
        }
      __syncthreads();
    }
  }
}

// ---------------- sequential chunk scan; block = (batch, v-tile of 64 cols) ----------------
__global__ void delta_recurrence(const bf16* __restrict__ Qp, const bf16* __restrict__ Kp,
                                 const bf16* __restrict__ Wp, const bf16* __restrict__ Up,
                                 bf16* __restrict__ Op) {
  extern __shared__ char smem[];
  const int SKS = 520;
  const int T64 = 64 * 72;                // one 64x64 padded tile
  bf16* St  = (bf16*)smem;                // [64 v][520 k]  S^T slice, resident
  bf16* WsB = St + 64 * SKS;              // 2 x [64][72]
  bf16* QsB = WsB + 2 * T64;              // 2 x [64][72]
  bf16* KsB = QsB + 2 * T64;              // 2 x [64][72]
  bf16* Kts = KsB + 2 * T64;              // [64][72]
  bf16* Ut  = Kts + T64;                  // u^T [v][c]
  bf16* Ab  = Ut + T64;                   // tril(Q K^T)
  const int tid = (int)threadIdx.x;
  const int wave = tid >> 5, wm = wave >> 1, wn = wave & 1;
  const int l = tid & 31, kh = l >> 4, l15 = l & 15;
  const int b = blockIdx.x >> 3, vt = blockIdx.x & 7, vbase = vt * 64;

  for (int i = tid; i < 64 * SKS; i += 256) St[i] = (bf16)0.f;
  __syncthreads();

  for (int n = 0; n < 64; ++n) {
    const size_t g0 = (size_t)b * 4096 + (size_t)n * 64;

    // ---- Phase A: u = U[:, vtile] - W @ S[:, vtile]  (double-buffered TDM)
    if (wave == 0) tdm_load_tile64(lds_off(WsB, smem), Wp + g0 * 512);
    v8f uacc[2] = {};
    for (int ks = 0; ks < 8; ++ks) {
      const int cur = ks & 1;
      const bf16* Wc = WsB + cur * T64;
      if (wave == 0) {
        if (ks + 1 < 8) {
          tdm_load_tile64(lds_off(WsB + (cur ^ 1) * T64, smem),
                          Wp + g0 * 512 + (ks + 1) * 64);
          tdm_wait1();
        } else tdm_wait0();
      }
      __syncthreads();
      #pragma unroll
      for (int k0 = 0; k0 < 64; k0 += 32) {
        v16bf a  = load_a_frag(Wc, 72, wm * 16, k0);
        v16bf b0 = load_b_frag(St, SKS, wn * 32, ks * 64 + k0);
        v16bf b1 = load_b_frag(St, SKS, wn * 32 + 16, ks * 64 + k0);
        uacc[0] = wmma_bf16(a, b0, uacc[0]);
        uacc[1] = wmma_bf16(a, b1, uacc[1]);
      }
      __syncthreads();
    }
    #pragma unroll
    for (int f = 0; f < 2; ++f)
      #pragma unroll
      for (int r = 0; r < 8; ++r) {
        int c = wm * 16 + r + 8 * kh;
        int v = wn * 32 + f * 16 + l15;
        float uval = (float)Up[(g0 + c) * 512 + vbase + v] - uacc[f][r];
        Ut[v * 72 + c] = (bf16)uval;
      }
    __syncthreads();

    // ---- Phase B+C1 merged: aacc = Q K^T, oacc = Q @ S  (Q/K tiles loaded once)
    if (wave == 0) {
      tdm_load_tile64(lds_off(QsB, smem), Qp + g0 * 512);
      tdm_load_tile64(lds_off(KsB, smem), Kp + g0 * 512);
    }
    v8f aacc[2] = {}, oacc[2] = {};
    for (int ks = 0; ks < 8; ++ks) {
      const int cur = ks & 1;
      const bf16* Qc = QsB + cur * T64;
      const bf16* Kc = KsB + cur * T64;
      if (wave == 0) {
        if (ks + 1 < 8) {
          tdm_load_tile64(lds_off(QsB + (cur ^ 1) * T64, smem),
                          Qp + g0 * 512 + (ks + 1) * 64);
          tdm_load_tile64(lds_off(KsB + (cur ^ 1) * T64, smem),
                          Kp + g0 * 512 + (ks + 1) * 64);
          tdm_wait2();
        } else tdm_wait0();
      }
      __syncthreads();
      #pragma unroll
      for (int k0 = 0; k0 < 64; k0 += 32) {
        v16bf a   = load_a_frag(Qc, 72, wm * 16, k0);
        v16bf kb0 = load_b_frag(Kc, 72, wn * 32, k0);
        v16bf kb1 = load_b_frag(Kc, 72, wn * 32 + 16, k0);
        v16bf sb0 = load_b_frag(St, SKS, wn * 32, ks * 64 + k0);
        v16bf sb1 = load_b_frag(St, SKS, wn * 32 + 16, ks * 64 + k0);
        aacc[0] = wmma_bf16(a, kb0, aacc[0]);
        aacc[1] = wmma_bf16(a, kb1, aacc[1]);
        oacc[0] = wmma_bf16(a, sb0, oacc[0]);
        oacc[1] = wmma_bf16(a, sb1, oacc[1]);
      }
      __syncthreads();
    }
    #pragma unroll
    for (int f = 0; f < 2; ++f)
      #pragma unroll
      for (int r = 0; r < 8; ++r) {
        int row = wm * 16 + r + 8 * kh;
        int col = wn * 32 + f * 16 + l15;
        Ab[row * 72 + col] = (bf16)(row >= col ? aacc[f][r] : 0.f);
      }
    __syncthreads();

    // ---- Phase C2: o += tril(Q K^T) @ u ; store o
    #pragma unroll
    for (int k0 = 0; k0 < 64; k0 += 32) {
      v16bf a  = load_a_frag(Ab, 72, wm * 16, k0);
      v16bf b0 = load_b_frag(Ut, 72, wn * 32, k0);
      v16bf b1 = load_b_frag(Ut, 72, wn * 32 + 16, k0);
      oacc[0] = wmma_bf16(a, b0, oacc[0]);
      oacc[1] = wmma_bf16(a, b1, oacc[1]);
    }
    #pragma unroll
    for (int f = 0; f < 2; ++f)
      #pragma unroll
      for (int r = 0; r < 8; ++r) {
        int c = wm * 16 + r + 8 * kh;
        int v = wn * 32 + f * 16 + l15;
        Op[(g0 + c) * 512 + vbase + v] = (bf16)oacc[f][r];
      }

    // ---- Phase D: S[:, vtile] += K^T @ u  (per 64-row k-segment, prefetched)
    if (wave == 0) tdm_load_tile64(lds_off(KsB, smem), Kp + g0 * 512);
    for (int ks = 0; ks < 8; ++ks) {
      const int cur = ks & 1;
      const bf16* Kc = KsB + cur * T64;
      if (wave == 0) {
        if (ks + 1 < 8) {
          tdm_load_tile64(lds_off(KsB + (cur ^ 1) * T64, smem),
                          Kp + g0 * 512 + (ks + 1) * 64);
          tdm_wait1();
        } else tdm_wait0();
      }
      __syncthreads();
      for (int i = tid; i < 64 * 64; i += 256) {   // Kts[kk][c] = Ks[c][kk]
        int c = i >> 6, kk = i & 63;
        Kts[kk * 72 + c] = Kc[c * 72 + kk];
      }
      __syncthreads();
      v8f sacc[2];
      #pragma unroll
      for (int f = 0; f < 2; ++f)
        #pragma unroll
        for (int r = 0; r < 8; ++r) {
          int krow = ks * 64 + wm * 16 + r + 8 * kh;
          int v = wn * 32 + f * 16 + l15;
          sacc[f][r] = (float)St[(size_t)v * SKS + krow];
        }
      #pragma unroll
      for (int k0 = 0; k0 < 64; k0 += 32) {
        v16bf a  = load_a_frag(Kts, 72, wm * 16, k0);
        v16bf b0 = load_b_frag(Ut, 72, wn * 32, k0);
        v16bf b1 = load_b_frag(Ut, 72, wn * 32 + 16, k0);
        sacc[0] = wmma_bf16(a, b0, sacc[0]);
        sacc[1] = wmma_bf16(a, b1, sacc[1]);
      }
      #pragma unroll
      for (int f = 0; f < 2; ++f)
        #pragma unroll
        for (int r = 0; r < 8; ++r) {
          int krow = ks * 64 + wm * 16 + r + 8 * kh;
          int v = wn * 32 + f * 16 + l15;
          St[(size_t)v * SKS + krow] = (bf16)sacc[f][r];
        }
      __syncthreads();
    }
  }
}

// ---------------- host orchestration ----------------
extern "C" void kernel_launch(void* const* d_in, const int* in_sizes, int n_in,
                              void* d_out, int out_size, void* d_ws, size_t ws_size,
                              hipStream_t stream) {
  (void)in_sizes; (void)n_in; (void)out_size; (void)ws_size;
  const float* X     = (const float*)d_in[0];
  const float* bq    = (const float*)d_in[2];
  const float* bk    = (const float*)d_in[4];
  const float* bv    = (const float*)d_in[6];
  const float* Wbeta = (const float*)d_in[7];
  const float* bbeta = (const float*)d_in[8];
  const float* bo    = (const float*)d_in[10];

  const size_t Mr = 8 * 4096, D = 512;
  char* ws = (char*)d_ws;
  size_t off = 0;
  auto take = [&](size_t bytes) { char* p = ws + off; off += (bytes + 255) & ~(size_t)255; return p; };
  bf16* Xb   = (bf16*)take(Mr * D * 2);
  bf16* Wqt  = (bf16*)take(D * D * 2);   // transposed bf16 weights [n][k]
  bf16* Wkt  = (bf16*)take(D * D * 2);
  bf16* Wvt  = (bf16*)take(D * D * 2);
  bf16* Wot  = (bf16*)take(D * D * 2);
  bf16* Qb   = (bf16*)take(Mr * D * 2);
  bf16* Kb   = (bf16*)take(Mr * D * 2);
  bf16* Vb   = (bf16*)take(Mr * D * 2);
  bf16* Wc   = (bf16*)take(Mr * D * 2);
  bf16* Uc   = (bf16*)take(Mr * D * 2);
  bf16* Ob   = (bf16*)take(Mr * D * 2);
  float* bta = (float*)take(Mr * 4);

  const size_t gsh = (size_t)(2 * 128 * 40 + 2 * 64 * 40) * sizeof(bf16);           // 30.7 KB
  const size_t csh = (size_t)2 * (64 * 520 * 2) + 64 * 65 * 4 + 2 * (64 * 72 * 2) + 256;
  const size_t rsh = (size_t)64 * 520 * 2 + 9 * (64 * 72 * 2);                      // 149.5 KB
  (void)hipFuncSetAttribute((const void*)chunkprep,
                            hipFuncAttributeMaxDynamicSharedMemorySize, (int)csh);
  (void)hipFuncSetAttribute((const void*)delta_recurrence,
                            hipFuncAttributeMaxDynamicSharedMemorySize, (int)rsh);

  dim3 tgrid((unsigned)(D / 64), (unsigned)(D / 64));
  cvt_f32_bf16<<<2048, 256, 0, stream>>>(X, Xb, (long)(Mr * D));
  cvt_transpose_bf16<<<tgrid, 256, 0, stream>>>((const float*)d_in[1], Wqt, (int)D);
  cvt_transpose_bf16<<<tgrid, 256, 0, stream>>>((const float*)d_in[3], Wkt, (int)D);
  cvt_transpose_bf16<<<tgrid, 256, 0, stream>>>((const float*)d_in[5], Wvt, (int)D);
  cvt_transpose_bf16<<<tgrid, 256, 0, stream>>>((const float*)d_in[9], Wot, (int)D);
  beta_kernel<<<Mr / 8, 256, 0, stream>>>(X, Wbeta, bbeta, bta, (int)D);

  dim3 ggrid((unsigned)(Mr / 128), (unsigned)(D / 64));
  gemm_wmma<false><<<ggrid, 256, gsh, stream>>>(Xb, Wqt, bq, Qb, (int)Mr, (int)D, (int)D);
  gemm_wmma<false><<<ggrid, 256, gsh, stream>>>(Xb, Wkt, bk, Kb, (int)Mr, (int)D, (int)D);
  gemm_wmma<false><<<ggrid, 256, gsh, stream>>>(Xb, Wvt, bv, Vb, (int)Mr, (int)D, (int)D);

  chunkprep<<<512, 256, csh, stream>>>(Kb, Vb, bta, Wc, Uc);
  delta_recurrence<<<64, 256, rsh, stream>>>(Qb, Kb, Wc, Uc, Ob);

  gemm_wmma<true><<<ggrid, 256, gsh, stream>>>(Ob, Wot, bo, d_out, (int)Mr, (int)D, (int)D);
}
